// primModel_23046794510701
// MI455X (gfx1250) — matrix operand order repaired and checked by reading, not compiled, not run on hardware
//
#include <hip/hip_runtime.h>
#include <hip/hip_bf16.h>

typedef __attribute__((ext_vector_type(16))) _Float16 v16h;
typedef __attribute__((ext_vector_type(8)))  _Float16 h8v;
typedef __attribute__((ext_vector_type(8)))  float    v8f;

#define KTAB 3
#define BATCH 512
#define DDIM 1024
#define HDIM 512
#define CCLS 48
#define TLEN 25
#define DP1 1025           // D+1
#define KX  1056           // D+1 padded to mult of 32
#define KFEAT 6288         // K*C + 2*K*D
#define KFP   6304         // padded
#define NLIN  1040         // 1025 padded to mult of 16 (65 tiles)

__device__ __forceinline__ v8f wmma16(v16h a, v16h b, v8f c) {
    return __builtin_amdgcn_wmma_f32_16x16x32_f16(false, a, false, b, (short)0, c, false, false);
}

// A fragment (16x32 f16) from row-major f16 (LDS or global), ld in halves.
// lane m = lane&15 is the row; half = lane>>4 selects K runs {8h..8h+7, 16+8h..16+8h+7}.
__device__ __forceinline__ v16h load_afrag(const _Float16* p0, int ld, int lane) {
    int m = lane & 15, hh = lane >> 4;
    const _Float16* p = p0 + (size_t)m * ld + 8 * hh;
    union { v16h v; h8v h[2]; } u;
    u.h[0] = *(const h8v*)(p);
    u.h[1] = *(const h8v*)(p + 16);
    return u.v;
}

// B fragment (32x16 f16) from pre-packed weights: lane holds 16 K halves, runs {8h..8h+7, 16+8h..16+8h+7}.
__device__ __forceinline__ v16h load_bfrag(const _Float16* pk, int Ntiles, int kb, int nb, int lane) {
    return *(const v16h*)(pk + (((size_t)kb * Ntiles + nb) * 32 + lane) * 16);
}

__device__ __forceinline__ float wred(float v) {
    for (int o = 16; o > 0; o >>= 1) v += __shfl_down(v, o, 32);
    return v;
}

__device__ __forceinline__ float sigm(float x) { return 1.f / (1.f + expf(-x)); }

// ---------------- weight packing: f32 -> f16 B-fragment layout ----------------
__global__ void pack_b_f16(const float* __restrict__ src, _Float16* __restrict__ dst,
                           int Kdim, int Nsrc, int Ntiles, int ld, int trans, int total) {
    int idx = blockIdx.x * 256 + threadIdx.x;
    if (idx >= total) return;
    int kr   = idx & 15;
    int lane = (idx >> 4) & 31;
    int blk  = idx >> 9;
    int nb   = blk % Ntiles;
    int kb   = blk / Ntiles;
    int n = nb * 16 + (lane & 15);
    int k = kb * 32 + (kr >> 3) * 16 + (lane >> 4) * 8 + (kr & 7);
    float v = 0.f;
    if (k < Kdim && n < Nsrc)
        v = trans ? src[(size_t)n * ld + k] : src[(size_t)k * ld + n];
    dst[idx] = (_Float16)v;
}

__global__ void cvt_f16_kernel(const float* __restrict__ src, _Float16* __restrict__ dst, int n) {
    int i = blockIdx.x * 256 + threadIdx.x;
    if (i < n) dst[i] = (_Float16)src[i];
}

// SR[k][b][0:1024]=S, [1024:2048]=R  (f16, classifier A operand)
__global__ void build_sr(const float* __restrict__ S, const float* __restrict__ R,
                         _Float16* __restrict__ SR) {
    size_t i = (size_t)blockIdx.x * 256 + threadIdx.x;
    if (i >= (size_t)KTAB * BATCH * 2 * DDIM) return;
    int d = (int)(i & 2047);
    size_t kb = i >> 11;
    float v = (d < DDIM) ? S[kb * DDIM + d] : R[kb * DDIM + (d - DDIM)];
    SR[i] = (_Float16)v;
}

// feat[b][144 + kk*1024 + d] = S[kk][b][d]; +3072 -> R; zero pad; Y part filled later
__global__ void build_feat(const float* __restrict__ S, const float* __restrict__ R,
                           _Float16* __restrict__ feat) {
    size_t i = (size_t)blockIdx.x * 256 + threadIdx.x;
    if (i >= (size_t)BATCH * KFP) return;
    int col = (int)(i % KFP);
    size_t b = i / KFP;
    if (col < KTAB * CCLS) return;  // classifier writes these
    float v = 0.f;
    if (col < 144 + 3 * DDIM) {
        int cc = col - 144; int kk = cc >> 10; int d = cc & 1023;
        v = S[((size_t)kk * BATCH + b) * DDIM + d];
    } else if (col < KFEAT) {
        int cc = col - 144 - 3 * DDIM; int kk = cc >> 10; int d = cc & 1023;
        v = R[((size_t)kk * BATCH + b) * DDIM + d];
    }
    feat[i] = (_Float16)v;
}

// ---------------- per-TAB classifier: softmax([S|R] @ cls_W + b) ----------------
__global__ __launch_bounds__(32) void classifier_kernel(
    const _Float16* __restrict__ SR, const _Float16* __restrict__ clspk,
    const float* __restrict__ cls_b, float* __restrict__ Y, _Float16* __restrict__ feat) {
    __shared__ float lg[16][48];
    int blk = blockIdx.x;
    int k = blk >> 5;          // TAB index
    int b0 = (blk & 31) * 16;  // batch tile
    int lane = threadIdx.x;
    const _Float16* A  = SR + ((size_t)k * BATCH + b0) * (2 * DDIM);
    const _Float16* pk = clspk + (size_t)k * (2 * DDIM) * CCLS;
    v8f acc[3];
    for (int nb = 0; nb < 3; nb++) {
        float bv = cls_b[k * CCLS + nb * 16 + (lane & 15)];
        for (int v = 0; v < 8; v++) acc[nb][v] = bv;
    }
    for (int kb = 0; kb < 64; kb++) {
        v16h a = load_afrag(A + kb * 32, 2 * DDIM, lane);
        for (int nb = 0; nb < 3; nb++)
            acc[nb] = wmma16(a, load_bfrag(pk, 3, kb, nb, lane), acc[nb]);
    }
    for (int nb = 0; nb < 3; nb++)
        for (int v = 0; v < 8; v++)
            lg[v + 8 * (lane >> 4)][nb * 16 + (lane & 15)] = acc[nb][v];
    __syncthreads();
    if (lane < 16) {
        int b = b0 + lane;
        float mx = -1e30f;
        for (int c = 0; c < 48; c++) mx = fmaxf(mx, lg[lane][c]);
        float s = 0.f;
        for (int c = 0; c < 48; c++) s += expf(lg[lane][c] - mx);
        float inv = 1.f / s;
        for (int c = 0; c < 48; c++) {
            float y = expf(lg[lane][c] - mx) * inv;
            Y[((size_t)k * BATCH + b) * CCLS + c] = y;
            feat[(size_t)b * KFP + k * CCLS + c] = (_Float16)y;
        }
    }
}

__global__ void curr_action_kernel(const float* __restrict__ Y, float* __restrict__ out) {
    int i = blockIdx.x * 256 + threadIdx.x;
    if (i >= BATCH * CCLS) return;
    out[i] = Y[i] + Y[BATCH * CCLS + i] + Y[2 * BATCH * CCLS + i];
}

// ---------------- x0 = feat @ lin_W + lin_b ----------------
__global__ __launch_bounds__(256) void x0_kernel(
    const _Float16* __restrict__ feat, const _Float16* __restrict__ linpk,
    const float* __restrict__ lin_b, float* __restrict__ x0) {
    int b0 = blockIdx.x * 16;
    int w = threadIdx.x >> 5, lane = threadIdx.x & 31;
    const _Float16* A = feat + (size_t)b0 * KFP;
    v8f acc[9];
    int nt[9], cnt = 0;
    for (int nb = w; nb < 65; nb += 8) nt[cnt++] = nb;
    for (int i = 0; i < 9; i++) {
        if (i >= cnt) break;
        int n = nt[i] * 16 + (lane & 15);
        float bv = (n < DP1) ? lin_b[n] : 0.f;
        for (int v = 0; v < 8; v++) acc[i][v] = bv;
    }
    for (int kb = 0; kb < KFP / 32; kb++) {
        v16h a = load_afrag(A + kb * 32, KFP, lane);
        #pragma unroll
        for (int i = 0; i < 9; i++) {
            if (i >= cnt) break;
            acc[i] = wmma16(a, load_bfrag(linpk, 65, kb, nt[i], lane), acc[i]);
        }
    }
    for (int i = 0; i < 9; i++) {
        if (i >= cnt) break;
        int n = nt[i] * 16 + (lane & 15);
        if (n < DP1)
            for (int v = 0; v < 8; v++)
                x0[(size_t)(b0 + v + 8 * (lane >> 4)) * DP1 + n] = acc[i][v];
    }
}

// ---------------- curr_dur + step-invariant attention-dur dot ----------------
__global__ __launch_bounds__(32) void small_kernel(
    const float* __restrict__ S, const float* __restrict__ R,
    const float* __restrict__ dur_W, const float* __restrict__ dur_b,
    const float* __restrict__ Wdur, float* __restrict__ out_dur, float* __restrict__ sdw) {
    int b = blockIdx.x, lane = threadIdx.x;
    float s = 0.f;
    for (int i = lane; i < KTAB * DDIM; i += 32) {
        int kk = i >> 10, d = i & 1023;
        s += R[((size_t)kk * BATCH + b) * DDIM + d] * dur_W[i];
    }
    s = wred(s);
    if (lane == 0) out_dur[(size_t)b * (TLEN + 1)] = s + dur_b[0];
    for (int k = 0; k < KTAB; k++) {
        float t = 0.f;
        for (int d = lane; d < DDIM; d += 32)
            t += S[((size_t)k * BATCH + b) * DDIM + d] * Wdur[d];
        t = wred(t);
        if (lane == 0) sdw[b * KTAB + k] = t;
    }
}

// ---------------- persistent LSTM decoder: 1 workgroup per 16 batch rows ----------------
__global__ __launch_bounds__(256) void decoder_kernel(
    const _Float16* __restrict__ Wih_pk, const _Float16* __restrict__ Whh_pk,
    const _Float16* __restrict__ Wp_pk,  const _Float16* __restrict__ Wattn_pk,
    const float* __restrict__ b_ih, const float* __restrict__ b_hh,
    const float* __restrict__ bp,   const float* __restrict__ Wdur,
    const float* __restrict__ bdur, const _Float16* __restrict__ embed16,
    const float* __restrict__ S,    const float* __restrict__ x0,
    const float* __restrict__ sdw,
    float* __restrict__ out_labels, float* __restrict__ out_probs,
    float* __restrict__ out_dur,    float* __restrict__ out_att) {
    __shared__ _Float16 xq[16 * KX];      // x (gates input), later reused for q
    __shared__ _Float16 hbuf[16 * HDIM];  // hidden state, f16
    __shared__ float lg[16 * 48];
    __shared__ float sc[16 * 3];
    __shared__ float aw[16 * 3];
    __shared__ float hdur[16];
    __shared__ float dur16[16];
    __shared__ int   lbl[16];

    int b0 = blockIdx.x * 16;
    int tid = threadIdx.x;
    int w = tid >> 5, lane = tid & 31;

    for (int i = tid; i < 16 * KX; i += 256) {
        int m = i / KX, col = i % KX;
        float v = (col < DP1) ? x0[(size_t)(b0 + m) * DP1 + col] : 0.f;
        xq[i] = (_Float16)v;
    }
    for (int i = tid; i < 16 * HDIM; i += 256) hbuf[i] = (_Float16)0.f;
    v8f cst[4];
    for (int s = 0; s < 4; s++) for (int v = 0; v < 8; v++) cst[s][v] = 0.f;
    __syncthreads();

    const float inv_sqrt_d = 0.03125f;  // 1/sqrt(1024)

    for (int t = 0; t < TLEN; t++) {
        // (a) hdur[b] = prev_h . Wdur[D:]   (h entering this step)
        for (int r = 0; r < 2; r++) {
            int m = 2 * w + r;
            float s = 0.f;
            for (int j = lane; j < HDIM; j += 32)
                s += (float)hbuf[m * HDIM + j] * Wdur[DDIM + j];
            s = wred(s);
            if (lane == 0) hdur[m] = s;
        }
        // (b) gates: wave w owns hidden units [64w, 64w+64): 16 tiles covering all 4 gates
        v8f acc[16];
        for (int g = 0; g < 4; g++)
            for (int s2 = 0; s2 < 4; s2++) {
                int col = g * HDIM + 64 * w + s2 * 16 + (lane & 15);
                float bv = b_ih[col] + b_hh[col];
                for (int v = 0; v < 8; v++) acc[g * 4 + s2][v] = bv;
            }
        for (int kb = 0; kb < KX / 32; kb++) {
            v16h a = load_afrag(xq + kb * 32, KX, lane);
            #pragma unroll
            for (int g = 0; g < 4; g++)
                for (int s2 = 0; s2 < 4; s2++) {
                    int nb = g * 32 + 4 * w + s2;
                    acc[g * 4 + s2] = wmma16(a, load_bfrag(Wih_pk, 128, kb, nb, lane), acc[g * 4 + s2]);
                }
        }
        for (int kb = 0; kb < HDIM / 32; kb++) {
            v16h a = load_afrag(hbuf + kb * 32, HDIM, lane);
            #pragma unroll
            for (int g = 0; g < 4; g++)
                for (int s2 = 0; s2 < 4; s2++) {
                    int nb = g * 32 + 4 * w + s2;
                    acc[g * 4 + s2] = wmma16(a, load_bfrag(Whh_pk, 128, kb, nb, lane), acc[g * 4 + s2]);
                }
        }
        __syncthreads();
        // (c) LSTM cell update, c in registers, new h -> LDS f16
        for (int s2 = 0; s2 < 4; s2++) {
            for (int v = 0; v < 8; v++) {
                float it = sigm(acc[0 * 4 + s2][v]);
                float ft = sigm(acc[1 * 4 + s2][v]);
                float gt = tanhf(acc[2 * 4 + s2][v]);
                float ot = sigm(acc[3 * 4 + s2][v]);
                float cn = ft * cst[s2][v] + it * gt;
                cst[s2][v] = cn;
                float hn = ot * tanhf(cn);
                int m = v + 8 * (lane >> 4);
                int col = 64 * w + s2 * 16 + (lane & 15);
                hbuf[m * HDIM + col] = (_Float16)hn;
            }
        }
        __syncthreads();
        // (d) logits (waves 0-2) and q = h @ Wattn (all waves); q overwrites x region
        if (w < 3) {
            v8f la;
            float bv = bp[w * 16 + (lane & 15)];
            for (int v = 0; v < 8; v++) la[v] = bv;
            for (int kb = 0; kb < HDIM / 32; kb++) {
                v16h a = load_afrag(hbuf + kb * 32, HDIM, lane);
                la = wmma16(a, load_bfrag(Wp_pk, 3, kb, w, lane), la);
            }
            for (int v = 0; v < 8; v++)
                lg[(v + 8 * (lane >> 4)) * 48 + w * 16 + (lane & 15)] = la[v];
        }
        {
            v8f qa[8];
            for (int i = 0; i < 8; i++) for (int v = 0; v < 8; v++) qa[i][v] = 0.f;
            for (int kb = 0; kb < HDIM / 32; kb++) {
                v16h a = load_afrag(hbuf + kb * 32, HDIM, lane);
                #pragma unroll
                for (int i = 0; i < 8; i++)
                    qa[i] = wmma16(a, load_bfrag(Wattn_pk, 64, kb, w * 8 + i, lane), qa[i]);
            }
            for (int i = 0; i < 8; i++) {
                int nb = w * 8 + i;
                for (int v = 0; v < 8; v++) {
                    int m = v + 8 * (lane >> 4);
                    xq[m * KX + nb * 16 + (lane & 15)] = (_Float16)(qa[i][v] * inv_sqrt_d);
                }
            }
        }
        __syncthreads();
        // (e) argmax + softmax of logits
        if (w == 0 && lane < 16) {
            int m = lane, b = b0 + m;
            float mx = -1e30f; int am = 0;
            for (int c = 0; c < 48; c++) { float v = lg[m * 48 + c]; if (v > mx) { mx = v; am = c; } }
            float ssum = 0.f;
            for (int c = 0; c < 48; c++) ssum += expf(lg[m * 48 + c] - mx);
            float inv = 1.f / ssum;
            for (int c = 0; c < 48; c++)
                out_probs[((size_t)b * TLEN + t) * CCLS + c] = expf(lg[m * 48 + c] - mx) * inv;
            out_labels[(size_t)t * BATCH + b] = (float)am;
            lbl[m] = am;
        }
        // (f) attention scores: 48 (row,k) pairs, 6 per wave
        for (int i = 0; i < 6; i++) {
            int p = w * 6 + i, m = p / 3, k = p % 3;
            const float* Sp = S + ((size_t)k * BATCH + b0 + m) * DDIM;
            float s = 0.f;
            for (int d = lane; d < DDIM; d += 32)
                s += (float)xq[m * KX + d] * Sp[d];
            s = wred(s);
            if (lane == 0) sc[m * 3 + k] = s;
        }
        __syncthreads();
        // softmax over K=3 + duration
        if (w == 0 && lane < 16) {
            int m = lane, b = b0 + m;
            float m0 = fmaxf(sc[m * 3], fmaxf(sc[m * 3 + 1], sc[m * 3 + 2]));
            float e0 = expf(sc[m * 3] - m0), e1 = expf(sc[m * 3 + 1] - m0), e2 = expf(sc[m * 3 + 2] - m0);
            float inv = 1.f / (e0 + e1 + e2);
            aw[m * 3] = e0 * inv; aw[m * 3 + 1] = e1 * inv; aw[m * 3 + 2] = e2 * inv;
            float dur = aw[m * 3] * sdw[b * 3] + aw[m * 3 + 1] * sdw[b * 3 + 1]
                      + aw[m * 3 + 2] * sdw[b * 3 + 2] + hdur[m] + bdur[0];
            dur16[m] = dur;
            out_dur[(size_t)b * (TLEN + 1) + t + 1] = dur;
        }
        __syncthreads();
        // (g) att output + rebuild x = [embed[label], dur]
        for (int e = tid; e < 16 * DDIM; e += 256) {
            int m = e >> 10, d = e & 1023, b = b0 + m;
            float att = aw[m * 3]     * S[((size_t)0 * BATCH + b) * DDIM + d]
                      + aw[m * 3 + 1] * S[((size_t)1 * BATCH + b) * DDIM + d]
                      + aw[m * 3 + 2] * S[((size_t)2 * BATCH + b) * DDIM + d];
            out_att[((size_t)t * BATCH + b) * DDIM + d] = att;
            xq[m * KX + d] = embed16[(size_t)lbl[m] * DDIM + d];
        }
        if (tid < 16) xq[tid * KX + DDIM] = (_Float16)dur16[tid];
        __syncthreads();
    }
}

extern "C" void kernel_launch(void* const* d_in, const int* in_sizes, int n_in,
                              void* d_out, int out_size, void* d_ws, size_t ws_size,
                              hipStream_t stream) {
    const float* S      = (const float*)d_in[0];
    const float* R      = (const float*)d_in[1];
    const float* cls_W  = (const float*)d_in[2];
    const float* cls_b  = (const float*)d_in[3];
    const float* dur_W  = (const float*)d_in[4];
    const float* dur_b  = (const float*)d_in[5];
    const float* lin_W  = (const float*)d_in[6];
    const float* lin_b  = (const float*)d_in[7];
    const float* W_ih   = (const float*)d_in[8];
    const float* W_hh   = (const float*)d_in[9];
    const float* b_ih   = (const float*)d_in[10];
    const float* b_hh   = (const float*)d_in[11];
    const float* Wp     = (const float*)d_in[12];
    const float* bp     = (const float*)d_in[13];
    const float* Wdur   = (const float*)d_in[14];
    const float* bdur   = (const float*)d_in[15];
    const float* embed  = (const float*)d_in[16];
    const float* Wattn  = (const float*)d_in[17];

    char* ws = (char*)d_ws;
    size_t off = 0;
    auto ah = [&](size_t n) { char* p = ws + off; off += ((n * 2 + 255) / 256) * 256; return (_Float16*)p; };
    auto af = [&](size_t n) { char* p = ws + off; off += ((n * 4 + 255) / 256) * 256; return (float*)p; };

    _Float16* Wih_pk   = ah((size_t)KX * 2048);
    _Float16* Whh_pk   = ah((size_t)HDIM * 2048);
    _Float16* Wp_pk    = ah((size_t)HDIM * 48);
    _Float16* Wattn_pk = ah((size_t)HDIM * DDIM);
    _Float16* lin_pk   = ah((size_t)KFP * NLIN);
    _Float16* cls_pk   = ah((size_t)KTAB * 2048 * 48);
    _Float16* embed16  = ah((size_t)CCLS * DDIM);
    _Float16* SR16     = ah((size_t)KTAB * BATCH * 2048);
    _Float16* feat16   = ah((size_t)BATCH * KFP);
    float*    Ybuf     = af((size_t)KTAB * BATCH * CCLS);
    float*    x0buf    = af((size_t)BATCH * DP1);
    float*    sdwbuf   = af((size_t)BATCH * KTAB);

    float* out = (float*)d_out;
    float* out_labels = out;                               // [T,B]
    float* out_probs  = out + (size_t)TLEN * BATCH;        // [B,T,C]
    float* out_act    = out_probs + (size_t)BATCH * TLEN * CCLS;   // [B,C]
    float* out_dur    = out_act + (size_t)BATCH * CCLS;            // [B,T+1]
    float* out_att    = out_dur + (size_t)BATCH * (TLEN + 1);      // [T,B,D]

    auto g = [](int total) { return dim3((total + 255) / 256); };

    // weight packing
    { int tot = KX * 2048;       pack_b_f16<<<g(tot), 256, 0, stream>>>(W_ih,  Wih_pk,   DP1,  2048, 128, DP1,  1, tot); }
    { int tot = HDIM * 2048;     pack_b_f16<<<g(tot), 256, 0, stream>>>(W_hh,  Whh_pk,   HDIM, 2048, 128, HDIM, 1, tot); }
    { int tot = HDIM * 48;       pack_b_f16<<<g(tot), 256, 0, stream>>>(Wp,    Wp_pk,    HDIM, 48,   3,   48,   0, tot); }
    { int tot = HDIM * DDIM;     pack_b_f16<<<g(tot), 256, 0, stream>>>(Wattn, Wattn_pk, HDIM, DDIM, 64,  DDIM, 0, tot); }
    { int tot = KFP * NLIN;      pack_b_f16<<<g(tot), 256, 0, stream>>>(lin_W, lin_pk,   KFEAT, DP1, 65,  DP1,  0, tot); }
    for (int k = 0; k < KTAB; k++) {
        int tot = 2048 * 48;
        pack_b_f16<<<g(tot), 256, 0, stream>>>(cls_W + (size_t)k * 2048 * 48,
                                               cls_pk + (size_t)k * 2048 * 48,
                                               2048, 48, 3, 48, 0, tot);
    }
    cvt_f16_kernel<<<g(CCLS * DDIM), 256, 0, stream>>>(embed, embed16, CCLS * DDIM);
    build_sr<<<g(KTAB * BATCH * 2048), 256, 0, stream>>>(S, R, SR16);
    build_feat<<<g(BATCH * KFP), 256, 0, stream>>>(S, R, feat16);

    classifier_kernel<<<dim3(KTAB * (BATCH / 16)), 32, 0, stream>>>(SR16, cls_pk, cls_b, Ybuf, feat16);
    curr_action_kernel<<<g(BATCH * CCLS), 256, 0, stream>>>(Ybuf, out_act);
    x0_kernel<<<dim3(BATCH / 16), 256, 0, stream>>>(feat16, lin_pk, lin_b, x0buf);
    small_kernel<<<dim3(BATCH), 32, 0, stream>>>(S, R, dur_W, dur_b, Wdur, out_dur, sdwbuf);

    decoder_kernel<<<dim3(BATCH / 16), 256, 0, stream>>>(
        Wih_pk, Whh_pk, Wp_pk, Wattn_pk, b_ih, b_hh, bp, Wdur, bdur,
        embed16, S, x0buf, sdwbuf,
        out_labels, out_probs, out_dur, out_att);
}
